// JSBF_84937273245790
// MI455X (gfx1250) — matrix operand order repaired and checked by reading, not compiled, not run on hardware
//
#include <hip/hip_runtime.h>
#include <stdint.h>

// ---------------- problem constants ----------------
#define D_X   256
#define D_Y   32
#define D_XY  288
#define HID   256
#define D_IN  800
#define NROW  4096
#define TSTEPS 50

// LDS strides (elements), 16B-aligned and bank-conflict-free for b128 A loads.
#define HSTR  808
#define H1STR 264

typedef __attribute__((ext_vector_type(16))) __bf16 v16bf;
typedef __attribute__((ext_vector_type(8)))  __bf16 v8bf;
typedef __attribute__((ext_vector_type(8)))  float  v8f;
typedef __bf16 bf16_t;

union ABf { v16bf v; v8bf h[2]; };

// Scheduling fence: keeps the k+1 prefetch loads from being sunk below the
// k-iteration WMMA group (round-3 codegen collapsed the ping-pong buffer).
#if __has_builtin(__builtin_amdgcn_sched_barrier)
#define SCHED_FENCE() __builtin_amdgcn_sched_barrier(0)
#else
#define SCHED_FENCE()
#endif

__device__ __forceinline__ bf16_t f2bf(float f) {
  union { float f; uint32_t u; } x; x.f = f;
  uint32_t r = x.u + 0x7FFFu + ((x.u >> 16) & 1u);   // round-to-nearest-even
  union { uint16_t u; bf16_t b; } y; y.u = (uint16_t)(r >> 16);
  return y.b;
}
__device__ __forceinline__ float bf2f(bf16_t b) {
  union { bf16_t b; uint16_t u; } x; x.b = b;
  union { uint32_t u; float f; } y; y.u = ((uint32_t)x.u) << 16;
  return y.f;
}
__device__ __forceinline__ float silu_f(float x) { return x / (1.f + __expf(-x)); }

// Pointer laundering (blocks LICM promotion of L2-hot weights across the time
// loop -> round-1 scratch spills). address_space(1) casts keep global_load
// lowering (round-2 regressed to flat_load without them).
__device__ __forceinline__ const bf16_t* launder_bf(const bf16_t* p) {
  uintptr_t q = (uintptr_t)p;
  asm volatile("" : "+s"(q));
  return (const bf16_t*)q;
}
__device__ __forceinline__ const float* launder_f(const float* p) {
  uintptr_t q = (uintptr_t)p;
  asm volatile("" : "+s"(q));
  return (const float*)q;
}
__device__ __forceinline__ v8bf gld8(const bf16_t* p) {
  return *(__attribute__((address_space(1))) const v8bf*)(uintptr_t)p;
}
__device__ __forceinline__ float gldf(const float* p) {
  return *(__attribute__((address_space(1))) const float*)(uintptr_t)p;
}

// deterministic counter-based RNG (cannot match JAX threefry; deterministic per replay)
__device__ __forceinline__ uint32_t hash3(uint32_t a, uint32_t b, uint32_t c) {
  uint32_t x = a * 0x9E3779B9u ^ b * 0x85EBCA6Bu ^ c * 0xC2B2AE35u;
  x ^= x >> 16; x *= 0x7FEB352Du; x ^= x >> 15; x *= 0x846CA68Bu; x ^= x >> 16;
  return x;
}
__device__ __forceinline__ float u01(uint32_t h) { return (float)(h >> 8) * (1.0f/16777216.0f); }
__device__ __forceinline__ float gauss3(uint32_t a, uint32_t b, uint32_t c) {
  float u1 = u01(hash3(a, b, c*2u + 1u)) + 1e-9f;
  float u2 = u01(hash3(a, b, c*2u + 2u));
  return sqrtf(-2.f * __logf(u1)) * __cosf(6.28318530718f * u2);
}

// ---------------- software-pipelined WMMA tile helpers ----------------
// A: LDS row-major bf16 (stride ASTR elems); B: global col-major bf16 (stride BSTR).
// Ping-pong double buffering with a sched fence so loads of k+1 are pinned above
// the WMMA group of k; waits then cover a full k-iteration of latency.
template <int NKT, int ASTR, int BSTR>
__device__ __forceinline__ void gemm_w4(const bf16_t* Ab, const bf16_t* Bb,
                                        int mt, int ntb, int lrow, int lhalf,
                                        v8f acc[4]) {
  const bf16_t* ar = Ab + (mt * 16 + lrow) * ASTR + lhalf * 8;
  const bf16_t* bc[4];
  #pragma unroll
  for (int j = 0; j < 4; ++j) bc[j] = Bb + ((ntb + j) * 16 + lrow) * BSTR + lhalf * 8;
  ABf a[2]; ABf b[2][4];
  a[0].h[0] = *(const v8bf*)(ar);
  a[0].h[1] = *(const v8bf*)(ar + 16);
  #pragma unroll
  for (int j = 0; j < 4; ++j) { b[0][j].h[0] = gld8(bc[j]); b[0][j].h[1] = gld8(bc[j] + 16); }
  #pragma unroll
  for (int kt = 0; kt < NKT; ++kt) {
    const int cur = kt & 1, nxt = cur ^ 1;
    if (kt + 1 < NKT) {                      // issue k+1 loads before consuming k
      const int ko = (kt + 1) * 32;
      a[nxt].h[0] = *(const v8bf*)(ar + ko);
      a[nxt].h[1] = *(const v8bf*)(ar + ko + 16);
      #pragma unroll
      for (int j = 0; j < 4; ++j) {
        b[nxt][j].h[0] = gld8(bc[j] + ko);
        b[nxt][j].h[1] = gld8(bc[j] + ko + 16);
      }
    }
    SCHED_FENCE();                            // loads stay above, WMMAs below
    #pragma unroll
    for (int j = 0; j < 4; ++j)
      acc[j] = __builtin_amdgcn_wmma_f32_16x16x32_bf16(
          false, a[cur].v, false, b[cur][j].v, (short)0, acc[j], false, false);
    SCHED_FENCE();
  }
}

template <int NKT, int ASTR, int BSTR>
__device__ __forceinline__ v8f gemm_w1(const bf16_t* Ab, const bf16_t* Bb,
                                       int mt, int col, int lrow, int lhalf) {
  const bf16_t* ar = Ab + (mt * 16 + lrow) * ASTR + lhalf * 8;
  const bf16_t* bcp = Bb + col * BSTR + lhalf * 8;
  v8f acc = (v8f){0.f,0.f,0.f,0.f,0.f,0.f,0.f,0.f};
  ABf a[2]; ABf b[2];
  a[0].h[0] = *(const v8bf*)(ar);
  a[0].h[1] = *(const v8bf*)(ar + 16);
  b[0].h[0] = gld8(bcp);
  b[0].h[1] = gld8(bcp + 16);
  #pragma unroll
  for (int kt = 0; kt < NKT; ++kt) {
    const int cur = kt & 1, nxt = cur ^ 1;
    if (kt + 1 < NKT) {
      const int ko = (kt + 1) * 32;
      a[nxt].h[0] = *(const v8bf*)(ar + ko);
      a[nxt].h[1] = *(const v8bf*)(ar + ko + 16);
      b[nxt].h[0] = gld8(bcp + ko);
      b[nxt].h[1] = gld8(bcp + ko + 16);
    }
    SCHED_FENCE();
    acc = __builtin_amdgcn_wmma_f32_16x16x32_bf16(
        false, a[cur].v, false, b[cur].v, (short)0, acc, false, false);
    SCHED_FENCE();
  }
  return acc;
}

// ---------------- prep kernel 1: transpose + convert weights to bf16 ----------------
__global__ void convert_weights(const float* __restrict__ w1, const float* __restrict__ w2,
                                const float* __restrict__ shw, const float* __restrict__ te,
                                bf16_t* __restrict__ W1t, bf16_t* __restrict__ W2t,
                                bf16_t* __restrict__ SHt, bf16_t* __restrict__ TEb) {
  int idx = blockIdx.x * 256 + threadIdx.x;
  const int N1 = 256 * 800, N2 = 256 * 256, N3 = 288 * 256, N4 = 2 * 256;
  if (idx < N1) {
    int n = idx / 800, k = idx - n * 800;
    W1t[idx] = f2bf(w1[k * 256 + n]);                 // tr_w1 is (800,256)
  } else if (idx < N1 + N2) {
    int j = idx - N1; int n = j >> 8, k = j & 255;
    W2t[j] = f2bf(w2[k * 256 + n]);                   // tr_w2 is (256,256)
  } else if (idx < N1 + N2 + N3) {
    int j = idx - (N1 + N2); int n = j >> 8, k = j & 255;
    SHt[j] = f2bf(shw[k * 288 + n]);                  // sh_w is (256,288)
  } else if (idx < N1 + N2 + N3 + N4) {
    int j = idx - (N1 + N2 + N3);
    TEb[j] = f2bf(te[j]);                             // t_embed (2,256)
  }
}

// ---------------- prep kernel 2: time-MLP per step (tau is scalar per step) ----------------
__global__ void time_mlp_kernel(const float* __restrict__ w1, const float* __restrict__ b1v,
                                const float* __restrict__ w2, const float* __restrict__ b2v,
                                bf16_t* __restrict__ TMb) {
  __shared__ float s[256];
  int ti = blockIdx.x + 1;                            // 1..49
  float tau = (float)ti * (1.f / (float)TSTEPS);
  int j = threadIdx.x;
  s[j] = silu_f(tau * w1[j] + b1v[j]);
  __syncthreads();
  float acc = b2v[j];
  for (int k = 0; k < 256; ++k) acc += s[k] * w2[k * 256 + j];
  TMb[ti * 256 + j] = f2bf(acc);
}

// ---------------- main persistent kernel: one block = 32 rows, all 49 steps ----------------
__global__ __launch_bounds__(256)
void diffusion_main(const float* __restrict__ xy0, const int* __restrict__ t0,
                    const bf16_t* __restrict__ W1t, const bf16_t* __restrict__ W2t,
                    const bf16_t* __restrict__ SHt, const bf16_t* __restrict__ TEb,
                    const bf16_t* __restrict__ TMb,
                    const float* __restrict__ b1, const float* __restrict__ b2,
                    const float* __restrict__ shb,
                    const float* __restrict__ chw, const float* __restrict__ chb,
                    float* __restrict__ out) {
  extern __shared__ char smem[];
  float*  xyf = (float*)smem;                                          // 32*288 f32
  bf16_t* Hs  = (bf16_t*)(smem + 32*D_XY*4);                           // 32*HSTR bf16
  bf16_t* H1  = (bf16_t*)(smem + 32*D_XY*4 + 32*HSTR*2);               // 32*H1STR bf16
  bf16_t* H2  = (bf16_t*)(smem + 32*D_XY*4 + 32*HSTR*2 + 32*H1STR*2);  // 32*H1STR bf16
  int*    Ts  = (int*)   (smem + 32*D_XY*4 + 32*HSTR*2 + 2*32*H1STR*2);

  const int tid   = threadIdx.x;
  const int wave  = tid >> 5;
  const int lane  = tid & 31;
  const int lhalf = lane >> 4;       // which K-half this lane holds
  const int lrow  = lane & 15;       // row (A) or column (B/C/D) within tile
  const int row0  = blockIdx.x * 32;

  for (int i = tid; i < 32 * D_XY; i += 256) xyf[i] = xy0[row0 * D_XY + i];
  if (tid < 32) Ts[tid] = t0[row0 + tid];
  __syncthreads();

  for (int ti = TSTEPS - 1; ti >= 1; --ti) {
    // Fresh opaque weight bases each step: blocks cross-iteration load promotion.
    const bf16_t* W1p = launder_bf(W1t);
    const bf16_t* W2p = launder_bf(W2t);
    const bf16_t* SHp = launder_bf(SHt);
    const float*  CHp = launder_f(chw);

    const float tau  = (float)ti       * (1.f / (float)TSTEPS);
    const float taup = (float)(ti - 1) * (1.f / (float)TSTEPS);
    const float sig  = 0.002f * __expf(tau  * 6.21460809842219f);   // ln(500)
    const float sigp = 0.002f * __expf(taup * 6.21460809842219f);
    const float stepsz = sig * sig - sigp * sigp;
    const float sqst   = sqrtf(stepsz);

    // ---- build h = [xy | te[t] | tm(step)] as bf16 in LDS ----
    for (int i = tid; i < 32 * D_XY; i += 256) {
      int r = i / D_XY, c = i - r * D_XY;
      Hs[r * HSTR + c] = f2bf(xyf[i]);
    }
    for (int i = tid; i < 32 * HID; i += 256) {
      int r = i >> 8, c = i & 255;
      Hs[r * HSTR + D_XY + c] = TEb[Ts[r] * HID + c];
    }
    for (int i = tid; i < 32 * HID; i += 256) {
      int r = i >> 8, c = i & 255;
      Hs[r * HSTR + D_XY + HID + c] = TMb[ti * HID + c];
    }
    // prefetch next step's time-embedding row into cache (global_prefetch_b8)
    if (ti > 1 && tid < 4)
      __builtin_prefetch(TMb + (ti - 1) * HID + tid * 64, 0, 3);
    __syncthreads();

    // ---- GEMM1: H1 = silu(Hs @ W1 + b1); wave = 1 row-tile x 4 col-tiles ----
    {
      const int mt  = wave >> 2;
      const int ntb = (wave & 3) * 4;
      v8f acc[4];
      #pragma unroll
      for (int j = 0; j < 4; ++j) acc[j] = (v8f){0.f,0.f,0.f,0.f,0.f,0.f,0.f,0.f};
      gemm_w4<D_IN/32, HSTR, D_IN>(Hs, W1p, mt, ntb, lrow, lhalf, acc);
      #pragma unroll
      for (int j = 0; j < 4; ++j) {
        const int col = (ntb + j) * 16 + lrow;
        const float bias = b1[col];
        #pragma unroll
        for (int v = 0; v < 8; ++v) {
          const int r = mt * 16 + lhalf * 8 + v;
          H1[r * H1STR + col] = f2bf(silu_f(acc[j][v] + bias));
        }
      }
    }
    __syncthreads();

    // ---- GEMM2: H2 = silu(H1 @ W2 + b2) ----
    {
      const int mt  = wave >> 2;
      const int ntb = (wave & 3) * 4;
      v8f acc[4];
      #pragma unroll
      for (int j = 0; j < 4; ++j) acc[j] = (v8f){0.f,0.f,0.f,0.f,0.f,0.f,0.f,0.f};
      gemm_w4<HID/32, H1STR, HID>(H1, W2p, mt, ntb, lrow, lhalf, acc);
      #pragma unroll
      for (int j = 0; j < 4; ++j) {
        const int col = (ntb + j) * 16 + lrow;
        const float bias = b2[col];
        #pragma unroll
        for (int v = 0; v < 8; ++v) {
          const int r = mt * 16 + lhalf * 8 + v;
          H2[r * H1STR + col] = f2bf(silu_f(acc[j][v] + bias));
        }
      }
    }
    __syncthreads();

    // ---- score head: xy += step * (H2 @ sh_w + sh_b) + sqrt(step) * N(0,1) ----
    for (int tile = wave; tile < 2 * 18; tile += 8) {
      const int mt = tile / 18, nt = tile - mt * 18;
      const int col = nt * 16 + lrow;
      v8f acc = gemm_w1<HID/32, H1STR, HID>(H2, SHp, mt, col, lrow, lhalf);
      const float bias = shb[col];
      #pragma unroll
      for (int v = 0; v < 8; ++v) {
        const int r = mt * 16 + lhalf * 8 + v;
        const float sc = acc[v] + bias;
        const float g  = gauss3((uint32_t)(row0 + r), (uint32_t)col, (uint32_t)ti);
        xyf[r * D_XY + col] += stepsz * sc + sqst * g;
      }
    }

    // ---- class head (K=2): Gumbel-argmax categorical ----
    if (tid < 32) {
      const int r = tid;
      float l0 = chb[0], l1 = chb[1];
      for (int k = 0; k < HID; ++k) {
        const float hv = bf2f(H2[r * H1STR + k]);
        l0 += hv * gldf(CHp + k * 2 + 0);
        l1 += hv * gldf(CHp + k * 2 + 1);
      }
      const float g0 = -__logf(-__logf(u01(hash3(row0 + r, 1000u, ti)) + 1e-9f) + 1e-9f);
      const float g1 = -__logf(-__logf(u01(hash3(row0 + r, 1001u, ti)) + 1e-9f) + 1e-9f);
      Ts[r] = (l1 + g1 > l0 + g0) ? 1 : 0;
    }
    __syncthreads();
  }

  // ---- writeback: x (4096x256), y (4096x32), t (4096) int bits ----
  for (int i = tid; i < 32 * D_X; i += 256) {
    int r = i >> 8, c = i & 255;
    out[(row0 + r) * D_X + c] = xyf[r * D_XY + c];
  }
  float* outy = out + (size_t)NROW * D_X;
  for (int i = tid; i < 32 * D_Y; i += 256) {
    int r = i >> 5, c = i & 31;
    outy[(row0 + r) * D_Y + c] = xyf[r * D_XY + D_X + c];
  }
  int* outt = (int*)(out + (size_t)NROW * D_XY);
  if (tid < 32) outt[row0 + tid] = Ts[tid];
}

// ---------------- host entry ----------------
extern "C" void kernel_launch(void* const* d_in, const int* in_sizes, int n_in,
                              void* d_out, int out_size, void* d_ws, size_t ws_size,
                              hipStream_t stream) {
  (void)in_sizes; (void)n_in; (void)out_size; (void)ws_size;
  const float* xy0   = (const float*)d_in[0];
  const int*   t0    = (const int*)  d_in[1];
  const float* t_emb = (const float*)d_in[2];
  const float* tm_w1 = (const float*)d_in[3];
  const float* tm_b1 = (const float*)d_in[4];
  const float* tm_w2 = (const float*)d_in[5];
  const float* tm_b2 = (const float*)d_in[6];
  const float* tr_w1 = (const float*)d_in[7];
  const float* tr_b1 = (const float*)d_in[8];
  const float* tr_w2 = (const float*)d_in[9];
  const float* tr_b2 = (const float*)d_in[10];
  const float* sh_w  = (const float*)d_in[11];
  const float* sh_b  = (const float*)d_in[12];
  const float* ch_w  = (const float*)d_in[13];
  const float* ch_b  = (const float*)d_in[14];

  char* ws = (char*)d_ws;
  bf16_t* W1t = (bf16_t*)(ws + 0);        // 256*800*2 = 409600
  bf16_t* W2t = (bf16_t*)(ws + 409600);   // 256*256*2 = 131072
  bf16_t* SHt = (bf16_t*)(ws + 540672);   // 288*256*2 = 147456
  bf16_t* TEb = (bf16_t*)(ws + 688128);   // 2*256*2   = 1024
  bf16_t* TMb = (bf16_t*)(ws + 689152);   // 50*256*2  = 25600

  const int convN = 256*800 + 256*256 + 288*256 + 2*256;
  convert_weights<<<(convN + 255) / 256, 256, 0, stream>>>(
      tr_w1, tr_w2, sh_w, t_emb, W1t, W2t, SHt, TEb);
  time_mlp_kernel<<<TSTEPS - 1, 256, 0, stream>>>(tm_w1, tm_b1, tm_w2, tm_b2, TMb);

  const size_t smem = (size_t)32*D_XY*4 + (size_t)32*HSTR*2 + 2*(size_t)32*H1STR*2 + 32*4;
  hipFuncSetAttribute(reinterpret_cast<const void*>(diffusion_main),
                      hipFuncAttributeMaxDynamicSharedMemorySize, (int)smem);
  diffusion_main<<<NROW / 32, 256, smem, stream>>>(
      xy0, t0, W1t, W2t, SHt, TEb, TMb, tr_b1, tr_b2, sh_b, ch_w, ch_b, (float*)d_out);
}
